// SingleHeadNeighborhoodAttentionModule_6923487281928
// MI455X (gfx1250) — matrix operand order repaired and checked by reading, not compiled, not run on hardware
//
#include <hip/hip_runtime.h>
#include <hip/hip_bf16.h>

// ---------------------------------------------------------------------------
// SingleHeadNeighborhoodAttention for MI455X (gfx1250, wave32, WMMA bf16)
//
//  * scores[b,k] = (Wq c_b . Wk n_bk)/8 = t_b . n_bk with M = Wq^T Wk / 8
//    precomputed once -> kills the 34-GFLOP k-projection.
//  * t = center x M and gate = [center|ctx] x Wg^T on v_wmma_f32_16x16x32_bf16.
//  * neigh (1 GB, the roofline cost) streamed via double-buffered
//    GLOBAL_LOAD_ASYNC_TO_LDS_B128 (ASYNCcnt) when the toolchain has the
//    builtins; synchronous float4 copy otherwise.
// ---------------------------------------------------------------------------

typedef __attribute__((ext_vector_type(16))) __bf16 v16bf;
typedef __attribute__((ext_vector_type(8)))  float  v8f;
typedef __attribute__((ext_vector_type(4)))  int    v4i;

// pointee types in explicit address spaces (per the builtin's signature:
// param0 = v4i in global AS, param1 = v4i in LDS AS)
typedef __attribute__((address_space(1))) v4i gv4i;
typedef __attribute__((address_space(3))) v4i lv4i;

#define EMB_DIM 256
#define ATTN_DIM 64
#define KNEI 16
#define WAVES 2   // waves (16-item tiles) per workgroup

#if __has_builtin(__builtin_amdgcn_global_load_async_to_lds_b128)
#define HAVE_ASYNC_LDS 1
#else
#define HAVE_ASYNC_LDS 0
#endif

#if HAVE_ASYNC_LDS
#if __has_builtin(__builtin_amdgcn_s_wait_asynccnt)
#define WAIT_ASYNC(n) __builtin_amdgcn_s_wait_asynccnt(n)
#else
#define WAIT_ASYNC(n) asm volatile("s_wait_asynccnt " #n ::: "memory")
#endif
#else
#define WAIT_ASYNC(n)
#endif

// ---------------- prepass 1: M = Wq^T Wk * (1/sqrt(64)) --------------------
__global__ void nh_compute_M(const float* __restrict__ Wq,
                             const float* __restrict__ Wk,
                             float* __restrict__ Mout) {
  int idx = blockIdx.x * blockDim.x + threadIdx.x;    // 256*256
  int i = idx >> 8;
  int j = idx & 255;
  float s = 0.f;
#pragma unroll 4
  for (int a = 0; a < ATTN_DIM; ++a)
    s += Wq[a * EMB_DIM + i] * Wk[a * EMB_DIM + j];
  Mout[idx] = s * 0.125f;
}

// ------------- prepass 2: pack M (256x256) into bf16 B-fragments -----------
// Fragment f = kt*16+nt holds B[kt*32 .. +31][nt*16 .. +15].
// Lane l, elem j: k = kt*32 + (l>=16 ? 8:0) + (j&7) + (j>=8 ? 16:0), n = nt*16 + (l&15).
__global__ void nh_pack_M(const float* __restrict__ M, __bf16* __restrict__ Mp) {
  int idx = blockIdx.x * blockDim.x + threadIdx.x;    // 8*16*512 = 65536
  int f    = idx >> 9;
  int lane = (idx >> 4) & 31;
  int j    = idx & 15;
  int kt = f >> 4, nt = f & 15;
  int k = kt * 32 + ((lane >> 4) << 3) + (j & 7) + ((j >> 3) << 4);
  int n = nt * 16 + (lane & 15);
  Mp[idx] = (__bf16)M[k * EMB_DIM + n];
}

// ------- prepass 3: pack Wg_w^T (512x256) into bf16 B-fragments ------------
__global__ void nh_pack_G(const float* __restrict__ Wg, __bf16* __restrict__ Gp) {
  int idx = blockIdx.x * blockDim.x + threadIdx.x;    // 16*16*512 = 131072
  int f    = idx >> 9;
  int lane = (idx >> 4) & 31;
  int j    = idx & 15;
  int kt = f >> 4, nt = f & 15;
  int k = kt * 32 + ((lane >> 4) << 3) + (j & 7) + ((j >> 3) << 4); // 0..511
  int n = nt * 16 + (lane & 15);                      // 0..255
  Gp[idx] = (__bf16)Wg[n * (2 * EMB_DIM) + k];        // B[k][n] = Wg_w[n][k]
}

// ---- helper: stream one 16x256 fp32 tile into LDS (async if available) ----
__device__ __forceinline__ void stream_tile(const float* __restrict__ np,
                                            float* __restrict__ dst, int lane) {
#pragma unroll 8
  for (int it = 0; it < 32; ++it) {
    int idx = (it * 32 + lane) * 4;                   // float offset, 16B units
#if HAVE_ASYNC_LDS
    __builtin_amdgcn_global_load_async_to_lds_b128(
        (gv4i*)(np + idx), (lv4i*)(dst + idx), 0, 0);
#else
    ((float4*)dst)[it * 32 + lane] = ((const float4*)np)[it * 32 + lane];
#endif
  }
}

// ------------------------------- main --------------------------------------
__global__ void __launch_bounds__(WAVES * 32)
nh_attn_main(const float* __restrict__ center,
             const float* __restrict__ neigh,
             const float* __restrict__ conf,
             const int*   __restrict__ valid,
             const float* __restrict__ bias,
             const __bf16* __restrict__ Mp,
             const __bf16* __restrict__ Gp,
             float* __restrict__ out) {
  extern __shared__ float smem[];
  const int lane = threadIdx.x & 31;
  const int w    = threadIdx.x >> 5;
  float* tbuf  = smem + w * (3 * 16 * EMB_DIM);       // t rows -> ctx rows
  float* nbufA = tbuf + 16 * EMB_DIM;                 // neigh ping
  float* nbufB = nbufA + 16 * EMB_DIM;                // neigh pong
  float* sc    = smem + WAVES * (3 * 16 * EMB_DIM) + w * 16;

  const long tile = (long)blockIdx.x * WAVES + w;     // 16 items per wave
  const long base = tile * 16;
  const float* nbase = neigh + base * KNEI * EMB_DIM;

  // kick off item 0's stream; hides first HBM latency under the t GEMM
  stream_tile(nbase, nbufA, lane);

  const int row  = lane & 15;                          // A-frag row (M index)
  const int koff = (lane >> 4) << 3;                   // +8 K-offset, hi lanes

  // ---- center A-fragments (fp32 -> bf16), reused by both GEMMs -----------
  v16bf caf[8];
  const float* crow = center + (base + row) * EMB_DIM;
#pragma unroll
  for (int kt = 0; kt < 8; ++kt) {
    float4 a0 = *(const float4*)(crow + kt * 32 + koff);
    float4 a1 = *(const float4*)(crow + kt * 32 + koff + 4);
    float4 a2 = *(const float4*)(crow + kt * 32 + 16 + koff);
    float4 a3 = *(const float4*)(crow + kt * 32 + 16 + koff + 4);
    caf[kt][0] = (__bf16)a0.x; caf[kt][1] = (__bf16)a0.y;
    caf[kt][2] = (__bf16)a0.z; caf[kt][3] = (__bf16)a0.w;
    caf[kt][4] = (__bf16)a1.x; caf[kt][5] = (__bf16)a1.y;
    caf[kt][6] = (__bf16)a1.z; caf[kt][7] = (__bf16)a1.w;
    caf[kt][8] = (__bf16)a2.x; caf[kt][9] = (__bf16)a2.y;
    caf[kt][10] = (__bf16)a2.z; caf[kt][11] = (__bf16)a2.w;
    caf[kt][12] = (__bf16)a3.x; caf[kt][13] = (__bf16)a3.y;
    caf[kt][14] = (__bf16)a3.z; caf[kt][15] = (__bf16)a3.w;
  }

  // ---- t = center(16x256) x M(256x256) -> LDS ----------------------------
  const int nloc = lane & 15;
  const int mo   = (lane < 16) ? 0 : 8;
  for (int nt = 0; nt < 16; ++nt) {
    v8f acc = {};
#pragma unroll
    for (int kt = 0; kt < 8; ++kt) {
      v16bf b = *(const v16bf*)(Mp + ((kt * 16 + nt) << 9) + (lane << 4));
      acc = __builtin_amdgcn_wmma_f32_16x16x32_bf16(
          false, caf[kt], false, b, (short)0, acc, false, false);
    }
#pragma unroll
    for (int r = 0; r < 8; ++r)
      tbuf[(mo + r) * EMB_DIM + nt * 16 + nloc] = acc[r];
  }

  // ---- per item: (async) stream neigh, scores, softmax, context ----------
  for (int i = 0; i < 16; ++i) {
    float* nb  = (i & 1) ? nbufB : nbufA;
    float* nbn = (i & 1) ? nbufA : nbufB;
    if (i < 15) {
      stream_tile(nbase + (i + 1) * KNEI * EMB_DIM, nbn, lane);
      WAIT_ASYNC(32);      // item i complete (in-order), i+1 in flight
    } else {
      WAIT_ASYNC(0);
    }

    // scores: lanes 0..15 each own one neighbor (float4 LDS reads)
    if (lane < 16) {
      const float4* t4 = (const float4*)(tbuf + i * EMB_DIM);
      const float4* n4 = (const float4*)(nb + lane * EMB_DIM);
      float s = 0.f;
#pragma unroll 8
      for (int d = 0; d < 64; ++d) {
        float4 ta = t4[d], na = n4[d];
        s += ta.x * na.x + ta.y * na.y + ta.z * na.z + ta.w * na.w;
      }
      sc[lane] = s + __logf(conf[(base + i) * KNEI + lane] + 1e-8f);
    }

    // softmax over 16 (redundant per lane; same-wave DS in-order)
    float mx = sc[0];
#pragma unroll
    for (int k2 = 1; k2 < 16; ++k2) mx = fmaxf(mx, sc[k2]);
    float a[16], den = 0.f;
#pragma unroll
    for (int k2 = 0; k2 < 16; ++k2) { a[k2] = __expf(sc[k2] - mx); den += a[k2]; }
    float inv = 1.f / den;

    // context: lane owns 8 cols (2 float4s); overwrite dead t-row i
    float4 c0 = {0.f, 0.f, 0.f, 0.f}, c1 = {0.f, 0.f, 0.f, 0.f};
    const float4* n4 = (const float4*)nb;
#pragma unroll
    for (int k2 = 0; k2 < 16; ++k2) {
      float4 b0 = n4[k2 * 64 + lane * 2];
      float4 b1 = n4[k2 * 64 + lane * 2 + 1];
      c0.x += a[k2] * b0.x; c0.y += a[k2] * b0.y;
      c0.z += a[k2] * b0.z; c0.w += a[k2] * b0.w;
      c1.x += a[k2] * b1.x; c1.y += a[k2] * b1.y;
      c1.z += a[k2] * b1.z; c1.w += a[k2] * b1.w;
    }
    c0.x *= inv; c0.y *= inv; c0.z *= inv; c0.w *= inv;
    c1.x *= inv; c1.y *= inv; c1.z *= inv; c1.w *= inv;
    ((float4*)(tbuf + i * EMB_DIM))[lane * 2]     = c0;
    ((float4*)(tbuf + i * EMB_DIM))[lane * 2 + 1] = c1;
  }

  // ---- ctx A-fragments (second half of gate K), float4 LDS reads ---------
  v16bf xaf[8];
  const float* xrow = tbuf + row * EMB_DIM;
#pragma unroll
  for (int kt = 0; kt < 8; ++kt) {
    float4 a0 = *(const float4*)(xrow + kt * 32 + koff);
    float4 a1 = *(const float4*)(xrow + kt * 32 + koff + 4);
    float4 a2 = *(const float4*)(xrow + kt * 32 + 16 + koff);
    float4 a3 = *(const float4*)(xrow + kt * 32 + 16 + koff + 4);
    xaf[kt][0] = (__bf16)a0.x; xaf[kt][1] = (__bf16)a0.y;
    xaf[kt][2] = (__bf16)a0.z; xaf[kt][3] = (__bf16)a0.w;
    xaf[kt][4] = (__bf16)a1.x; xaf[kt][5] = (__bf16)a1.y;
    xaf[kt][6] = (__bf16)a1.z; xaf[kt][7] = (__bf16)a1.w;
    xaf[kt][8] = (__bf16)a2.x; xaf[kt][9] = (__bf16)a2.y;
    xaf[kt][10] = (__bf16)a2.z; xaf[kt][11] = (__bf16)a2.w;
    xaf[kt][12] = (__bf16)a3.x; xaf[kt][13] = (__bf16)a3.y;
    xaf[kt][14] = (__bf16)a3.z; xaf[kt][15] = (__bf16)a3.w;
  }

  // ---- gate = [center|ctx](16x512) x Wg^T(512x256) + fused epilogue ------
  for (int nt = 0; nt < 16; ++nt) {
    v8f acc = {};
#pragma unroll
    for (int kt = 0; kt < 16; ++kt) {
      v16bf b = *(const v16bf*)(Gp + ((kt * 16 + nt) << 9) + (lane << 4));
      v16bf av = (kt < 8) ? caf[kt] : xaf[kt - 8];
      acc = __builtin_amdgcn_wmma_f32_16x16x32_bf16(
          false, av, false, b, (short)0, acc, false, false);
    }
    int col  = nt * 16 + nloc;
    float bc = bias[col];
#pragma unroll
    for (int r = 0; r < 8; ++r) {
      int m   = mo + r;
      float c = center[(base + m) * EMB_DIM + col];
      float x = tbuf[m * EMB_DIM + col];
      float g = 1.f / (1.f + __expf(-(acc[r] + bc)));
      float o = valid[base + m] ? (g * c + (1.f - g) * x) : c;
      out[(base + m) * EMB_DIM + col] = o;
    }
  }
}

// ---------------------------------------------------------------------------
extern "C" void kernel_launch(void* const* d_in, const int* in_sizes, int n_in,
                              void* d_out, int out_size, void* d_ws, size_t ws_size,
                              hipStream_t stream) {
  const float* center = (const float*)d_in[0];
  const float* neigh  = (const float*)d_in[1];
  const float* conf   = (const float*)d_in[2];
  const int*   valid  = (const int*)d_in[3];
  const float* Wq     = (const float*)d_in[4];
  const float* Wk     = (const float*)d_in[5];
  const float* Wg_w   = (const float*)d_in[6];
  const float* Wg_b   = (const float*)d_in[7];
  float* outp = (float*)d_out;

  const int B = in_sizes[0] / EMB_DIM;   // 65536

  float*  Mfp = (float*)d_ws;                                 // 256 KB
  __bf16* Mp  = (__bf16*)((char*)d_ws + 256 * 1024);          // 128 KB
  __bf16* Gp  = (__bf16*)((char*)d_ws + 384 * 1024);          // 256 KB

  nh_compute_M<<<(EMB_DIM * EMB_DIM) / 256, 256, 0, stream>>>(Wq, Wk, Mfp);
  nh_pack_M<<<65536 / 256, 256, 0, stream>>>(Mfp, Mp);
  nh_pack_G<<<131072 / 256, 256, 0, stream>>>(Wg_w, Gp);

  const int itemsPerWG = 16 * WAVES;                          // 32
  const int grid = B / itemsPerWG;                            // 2048
  const size_t shbytes = WAVES * (3 * 16 * EMB_DIM) * sizeof(float)
                       + WAVES * 16 * sizeof(float);          // ~96 KB -> 3 WG/WGP
  nh_attn_main<<<grid, WAVES * 32, shbytes, stream>>>(
      center, neigh, conf, valid, Wg_b, Mp, Gp, outp);
}
